// ChildSumTreeLSTMCell_15083925144171
// MI455X (gfx1250) — matrix operand order repaired and checked by reading, not compiled, not run on hardware
//
#include <hip/hip_runtime.h>
#include <hip/hip_bf16.h>

typedef __bf16 bf16;
typedef __attribute__((ext_vector_type(16))) __bf16 v16bf;
typedef __attribute__((ext_vector_type(8)))  float  v8f;
typedef __attribute__((ext_vector_type(4)))  int    int4v;

#define NNODE 16384
#define KCH   12
#define MROW  (NNODE*KCH)      // 196608
#define HDIM  256
#define D1IN  640              // H+ND+ED
#define D2    512              // H+ND

// ---------- address-space helpers (integer round-trip casts: always parse) ----------
typedef __attribute__((address_space(1))) int4v g_int4;
typedef __attribute__((address_space(3))) int4v l_int4;
__device__ __forceinline__ g_int4* to_global4(const void* p) {
    return (g_int4*)(unsigned long long)p;
}
__device__ __forceinline__ l_int4* to_lds4(const void* p) {
    // LDS offset = low 32 bits of the generic (flat) address per CDNA5 aperture rules
    return (l_int4*)(unsigned int)(unsigned long long)p;
}

#if defined(__HIP_DEVICE_COMPILE__) && defined(__has_builtin)
#if __has_builtin(__builtin_amdgcn_global_load_async_to_lds_b128) && \
    __has_builtin(__builtin_amdgcn_s_wait_asynccnt)
#define USE_ASYNC_LDS 1
#endif
#endif

#if defined(USE_ASYNC_LDS)
#define WAIT_ASYNC(n) __builtin_amdgcn_s_wait_asynccnt(n)
#else
#define WAIT_ASYNC(n) do { } while (0)
#endif

// ---------------- f32 -> bf16 convert/copy ----------------
__global__ void cvt_f32_bf16(const float* __restrict__ in, bf16* __restrict__ out, int n) {
    int i = blockIdx.x * 256 + threadIdx.x;
    if (i < n) out[i] = (bf16)in[i];
}

// ---------------- concat(h, embed, edge) -> bf16 [MROW x 640] ----------------
__global__ void pack_x_kernel(const float* __restrict__ h, const float* __restrict__ embed,
                              const float* __restrict__ edge, bf16* __restrict__ out,
                              long long total) {
    long long idx = (long long)blockIdx.x * 256 + threadIdx.x;
    if (idx >= total) return;
    long long row = idx / D1IN;
    int j = (int)(idx % D1IN);
    float v;
    if (j < 256)      v = h[row * 256 + j];
    else if (j < 512) v = embed[row * 256 + (j - 256)];
    else              v = edge[row * 128 + (j - 512)];
    out[idx] = (bf16)v;
}

// ---------------- bias_cat = [Wf_b+bf | Wi_b+bi | Wu_b+bu | Wo_b+bo] ----------------
__global__ void biascat_kernel(const float* wfb, const float* bf_, const float* wib, const float* bi_,
                               const float* wub, const float* bu_, const float* wob, const float* bo_,
                               float* __restrict__ out) {
    int i = blockIdx.x * 256 + threadIdx.x;
    if (i >= 1024) return;
    int seg = i >> 8, r = i & 255;
    const float* a = (seg == 0) ? wfb : (seg == 1) ? wib : (seg == 2) ? wub : wob;
    const float* b = (seg == 0) ? bf_ : (seg == 1) ? bi_ : (seg == 2) ? bu_ : bo_;
    out[i] = a[r] + b[r];
}

// ---------------- tiled bf16 WMMA GEMM: C[M,N] = act(A[M,K] * Bw[N,K]^T + bias) ----------------
// Bw is weight [N x K] row-major (K contiguous per output column).
// Double-buffered LDS; async global->LDS copies (ASYNCcnt) when available.
template <bool RELU, bool OUT_BF16>
__global__ __launch_bounds__(256)
void gemm_wmma_bf16(const bf16* __restrict__ A, const bf16* __restrict__ Bw,
                    const float* __restrict__ bias, void* __restrict__ Cout,
                    int M, int Nn, int Kd) {
    __shared__ bf16 lA[2][128 * 40];   // 128 M-rows x 32 K (+pad to 40 -> 80B rows, 16B aligned)
    __shared__ bf16 lB[2][128 * 40];   // 128 N-rows x 32 K (+pad)

    const int tid  = threadIdx.x;
    const int lane = tid & 31;
    const int wid  = tid >> 5;
    const int wm   = wid >> 2;                 // 0..1 -> 64-row slab
    const int wn   = wid & 3;                  // 0..3 -> 32-col slab
    const int m0   = blockIdx.y * 128;
    const int n0   = blockIdx.x * 128;

    v8f acc[4][2];
    #pragma unroll
    for (int i = 0; i < 4; ++i)
        #pragma unroll
        for (int j = 0; j < 2; ++j)
            #pragma unroll
            for (int v = 0; v < 8; ++v) acc[i][j][v] = 0.f;

    // staging geometry: wave w covers rows [w*16, w*16+16); each lane moves one 16B chunk.
    const int arow = wid * 16 + (lane >> 2);   // rows arow and arow+8
    const int acol = (lane & 3) * 8;           // bf16-element offset inside the 32-wide K slice

    auto stage_tiles = [&](int buf, int k0) {
        const bf16* ga0 = A  + (size_t)(m0 + arow)     * Kd + k0 + acol;
        const bf16* ga1 = A  + (size_t)(m0 + arow + 8) * Kd + k0 + acol;
        const bf16* gb0 = Bw + (size_t)(n0 + arow)     * Kd + k0 + acol;
        const bf16* gb1 = Bw + (size_t)(n0 + arow + 8) * Kd + k0 + acol;
        bf16* la0 = &lA[buf][(arow)     * 40 + acol];
        bf16* la1 = &lA[buf][(arow + 8) * 40 + acol];
        bf16* lb0 = &lB[buf][(arow)     * 40 + acol];
        bf16* lb1 = &lB[buf][(arow + 8) * 40 + acol];
#if defined(USE_ASYNC_LDS)
        __builtin_amdgcn_global_load_async_to_lds_b128(to_global4(ga0), to_lds4(la0), 0, 0);
        __builtin_amdgcn_global_load_async_to_lds_b128(to_global4(ga1), to_lds4(la1), 0, 0);
        __builtin_amdgcn_global_load_async_to_lds_b128(to_global4(gb0), to_lds4(lb0), 0, 0);
        __builtin_amdgcn_global_load_async_to_lds_b128(to_global4(gb1), to_lds4(lb1), 0, 0);
#else
        uint4 va0 = *(const uint4*)ga0;
        uint4 va1 = *(const uint4*)ga1;
        uint4 vb0 = *(const uint4*)gb0;
        uint4 vb1 = *(const uint4*)gb1;
        *(uint4*)la0 = va0;
        *(uint4*)la1 = va1;
        *(uint4*)lb0 = vb0;
        *(uint4*)lb1 = vb1;
#endif
    };

    const int nsteps = Kd >> 5;
    stage_tiles(0, 0);

    const int half = lane >> 4;
    const int r16  = lane & 15;

    for (int s = 0; s < nsteps; ++s) {
        const int cur = s & 1;
        if (s + 1 < nsteps) {
            stage_tiles(cur ^ 1, (s + 1) << 5);
            WAIT_ASYNC(4);                 // 4 oldest (current buffer) complete
        } else {
            WAIT_ASYNC(0);
        }
        __syncthreads();                   // all waves' tiles resident in lA/lB[cur]

        const bf16* bAc = lA[cur];
        const bf16* bBc = lB[cur];

        v16bf afr[4], bfr[2];
        #pragma unroll
        for (int tm = 0; tm < 4; ++tm) {
            // A 16x32 frag: lanes0-15 M=r16: V0..3=K[half*8..+7], V4..7=K[16+half*8..+7]
            const bf16* p = &bAc[(wm * 64 + tm * 16 + r16) * 40 + half * 8];
            union { uint4 u[2]; v16bf v; } cv;
            cv.u[0] = *(const uint4*)p;
            cv.u[1] = *(const uint4*)(p + 16);
            afr[tm] = cv.v;
        }
        #pragma unroll
        for (int tn = 0; tn < 2; ++tn) {
            // B 32x16 frag: column N=r16, V0..3=K[half*8..+7], V4..7=K[16+half*8..+7]
            const bf16* p = &bBc[(wn * 32 + tn * 16 + r16) * 40 + half * 8];
            union { uint4 u[2]; v16bf v; } cv;
            cv.u[0] = *(const uint4*)p;
            cv.u[1] = *(const uint4*)(p + 16);
            bfr[tn] = cv.v;
        }

        #pragma unroll
        for (int tm = 0; tm < 4; ++tm)
            #pragma unroll
            for (int tn = 0; tn < 2; ++tn)
                acc[tm][tn] = __builtin_amdgcn_wmma_f32_16x16x32_bf16(
                    false, afr[tm], false, bfr[tn], (short)0, acc[tm][tn], false, false);

        __syncthreads();                   // reads of [cur] done before it is restaged
    }

    // epilogue: C/D layout -> N = lane&15, M = v + 8*(lane>>4)
    const int mhalf = lane >> 4;
    const int ncol  = lane & 15;
    #pragma unroll
    for (int tm = 0; tm < 4; ++tm) {
        #pragma unroll
        for (int tn = 0; tn < 2; ++tn) {
            const int nc = n0 + wn * 32 + tn * 16 + ncol;
            const float bb = bias ? bias[nc] : 0.f;
            #pragma unroll
            for (int v = 0; v < 8; ++v) {
                const int mr = m0 + wm * 64 + tm * 16 + v + 8 * mhalf;
                float val = acc[tm][tn][v] + bb;
                if (RELU) val = val > 0.f ? val : 0.f;
                if (OUT_BF16) ((bf16*)Cout)[(size_t)mr * Nn + nc] = (bf16)val;
                else          ((float*)Cout)[(size_t)mr * Nn + nc] = val;
            }
        }
    }
}

// ---------------- per-node attention: scores, softmax, hbar, csum ----------------
// wave w owns columns [w*64, w*64+64); wave32 butterfly reductions for the 12 scores.
__global__ __launch_bounds__(256)
void attn_kernel(const bf16* __restrict__ H2, const float* __restrict__ cin,
                 const unsigned char* __restrict__ mask,
                 const float* __restrict__ attw, const float* __restrict__ attb,
                 bf16* __restrict__ hbar, float* __restrict__ csum) {
    __shared__ float wpart[8][KCH];
    __shared__ float at[KCH];
    const int n = blockIdx.x;
    const int t = threadIdx.x;
    const int w = t >> 5, lane = t & 31;
    const int c0 = w * 64 + lane;
    const int c1 = c0 + 32;
    const bf16* h2n = H2 + (size_t)n * KCH * D2;

    float hv0[KCH], hv1[KCH], part[KCH];
    const float aw0 = attw[c0], aw1 = attw[c1];
    #pragma unroll
    for (int k = 0; k < KCH; ++k) {
        float a = (float)h2n[k * D2 + c0];
        float b = (float)h2n[k * D2 + c1];
        hv0[k] = a; hv1[k] = b;
        part[k] = a * aw0 + b * aw1;
    }
    #pragma unroll
    for (int m = 16; m > 0; m >>= 1)
        #pragma unroll
        for (int k = 0; k < KCH; ++k)
            part[k] += __shfl_xor(part[k], m, 32);
    if (lane == 0)
        #pragma unroll
        for (int k = 0; k < KCH; ++k) wpart[w][k] = part[k];
    __syncthreads();

    if (t == 0) {
        float sc[KCH];
        for (int k = 0; k < KCH; ++k) {
            float s = 0.f;
            for (int ww = 0; ww < 8; ++ww) s += wpart[ww][k];
            sc[k] = s + attb[0];
        }
        float mx = -1e30f;
        for (int k = 0; k < KCH; ++k) if (mask[n * KCH + k] && sc[k] > mx) mx = sc[k];
        float s = 0.f;
        for (int k = 0; k < KCH; ++k) {
            float e = mask[n * KCH + k] ? __expf(sc[k] - mx) : 0.f;
            at[k] = e; s += e;
        }
        float inv = 1.f / s;
        for (int k = 0; k < KCH; ++k) at[k] *= inv;
    }
    __syncthreads();

    float o0 = 0.f, o1 = 0.f;
    #pragma unroll
    for (int k = 0; k < KCH; ++k) { o0 += at[k] * hv0[k]; o1 += at[k] * hv1[k]; }
    hbar[(size_t)n * D2 + c0] = (bf16)o0;
    hbar[(size_t)n * D2 + c1] = (bf16)o1;

    // csum over 256 hidden columns (one per thread)
    const float* cn = cin + (size_t)n * KCH * HDIM;
    float s = 0.f;
    #pragma unroll
    for (int k = 0; k < KCH; ++k) if (mask[n * KCH + k]) s += cn[k * HDIM + t];
    csum[(size_t)n * HDIM + t] = s;
}

// ---------------- LSTM gate epilogue ----------------
__global__ void final_kernel(const float* __restrict__ G, const float* __restrict__ csum,
                             float* __restrict__ out, int total) {
    int idx = blockIdx.x * 256 + threadIdx.x;
    if (idx >= total) return;
    int n = idx >> 8, j = idx & 255;
    const float* g = G + (size_t)n * 1024;
    float f = 1.f / (1.f + __expf(-g[j]));
    float i = 1.f / (1.f + __expf(-g[256 + j]));
    float u = tanhf(g[512 + j]);
    float o = 1.f / (1.f + __expf(-g[768 + j]));
    float cnew = i * u + f * csum[idx];
    float hnew = o * tanhf(cnew);
    out[idx]         = hnew;
    out[total + idx] = cnew;
}

// ---------------- workspace layout (bytes) ----------------
static const size_t XIN_OFF  = 0;                         // bf16 [MROW x 640]   (h2 aliases this)
static const size_t Z_OFF    = 251658240;                 // bf16 [MROW x 512]   (G aliases this)
static const size_t HBAR_OFF = 452984832;                 // bf16 [NNODE x 512]
static const size_t S1_OFF   = 469762048;                 // bf16 [NNODE x 512]
static const size_t S2_OFF   = 486539264;                 // bf16 [NNODE x 512]
static const size_t CSUM_OFF = 503316480;                 // f32  [NNODE x 256]
static const size_t EL1T_OFF = 520093696;                 // bf16 [512 x 640]
static const size_t EL2T_OFF = 520749056;                 // bf16 [512 x 512]
static const size_t NL1T_OFF = 521273344;                 // bf16 [512 x 512]
static const size_t NL2T_OFF = 521797632;                 // bf16 [512 x 512]
static const size_t WCAT_OFF = 522321920;                 // bf16 [1024 x 512]
static const size_t BCAT_OFF = 523370496;                 // f32  [1024]

extern "C" void kernel_launch(void* const* d_in, const int* in_sizes, int n_in,
                              void* d_out, int out_size, void* d_ws, size_t ws_size,
                              hipStream_t stream) {
    const float* h_in   = (const float*)d_in[0];
    const float* c_in   = (const float*)d_in[1];
    const float* embed  = (const float*)d_in[2];
    const float* edge   = (const float*)d_in[3];
    const unsigned char* mask = (const unsigned char*)d_in[4];
    const float* att_w  = (const float*)d_in[5];
    const float* att_b  = (const float*)d_in[6];
    const float* el1_w  = (const float*)d_in[7];
    const float* el1_b  = (const float*)d_in[8];
    const float* el2_w  = (const float*)d_in[9];
    const float* el2_b  = (const float*)d_in[10];
    const float* nl1_w  = (const float*)d_in[11];
    const float* nl1_b  = (const float*)d_in[12];
    const float* nl2_w  = (const float*)d_in[13];
    const float* nl2_b  = (const float*)d_in[14];
    const float* Wf_w = (const float*)d_in[15]; const float* Wf_b = (const float*)d_in[16]; const float* bf_ = (const float*)d_in[17];
    const float* Wi_w = (const float*)d_in[18]; const float* Wi_b = (const float*)d_in[19]; const float* bi_ = (const float*)d_in[20];
    const float* Wu_w = (const float*)d_in[21]; const float* Wu_b = (const float*)d_in[22]; const float* bu_ = (const float*)d_in[23];
    const float* Wo_w = (const float*)d_in[24]; const float* Wo_b = (const float*)d_in[25]; const float* bo_ = (const float*)d_in[26];

    char* ws = (char*)d_ws;
    bf16*  XIN  = (bf16*)(ws + XIN_OFF);
    bf16*  Z    = (bf16*)(ws + Z_OFF);
    bf16*  H2   = (bf16*)(ws + XIN_OFF);   // alias: XIN dead after GEMM1
    bf16*  HBAR = (bf16*)(ws + HBAR_OFF);
    bf16*  S1   = (bf16*)(ws + S1_OFF);
    bf16*  S2   = (bf16*)(ws + S2_OFF);
    float* CSUM = (float*)(ws + CSUM_OFF);
    float* G    = (float*)(ws + Z_OFF);    // alias: Z dead after GEMM2
    bf16*  EL1T = (bf16*)(ws + EL1T_OFF);
    bf16*  EL2T = (bf16*)(ws + EL2T_OFF);
    bf16*  NL1T = (bf16*)(ws + NL1T_OFF);
    bf16*  NL2T = (bf16*)(ws + NL2T_OFF);
    bf16*  WCAT = (bf16*)(ws + WCAT_OFF);
    float* BCAT = (float*)(ws + BCAT_OFF);

    // weight packs (f32 -> bf16; weights are already [N_out x K] row-major = B operand)
    cvt_f32_bf16<<<(512 * 640 + 255) / 256, 256, 0, stream>>>(el1_w, EL1T, 512 * 640);
    cvt_f32_bf16<<<(512 * 512 + 255) / 256, 256, 0, stream>>>(el2_w, EL2T, 512 * 512);
    cvt_f32_bf16<<<(512 * 512 + 255) / 256, 256, 0, stream>>>(nl1_w, NL1T, 512 * 512);
    cvt_f32_bf16<<<(512 * 512 + 255) / 256, 256, 0, stream>>>(nl2_w, NL2T, 512 * 512);
    cvt_f32_bf16<<<(256 * 512 + 255) / 256, 256, 0, stream>>>(Wf_w, WCAT + (size_t)0   * 512, 256 * 512);
    cvt_f32_bf16<<<(256 * 512 + 255) / 256, 256, 0, stream>>>(Wi_w, WCAT + (size_t)256 * 512, 256 * 512);
    cvt_f32_bf16<<<(256 * 512 + 255) / 256, 256, 0, stream>>>(Wu_w, WCAT + (size_t)512 * 512, 256 * 512);
    cvt_f32_bf16<<<(256 * 512 + 255) / 256, 256, 0, stream>>>(Wo_w, WCAT + (size_t)768 * 512, 256 * 512);
    biascat_kernel<<<4, 256, 0, stream>>>(Wf_b, bf_, Wi_b, bi_, Wu_b, bu_, Wo_b, bo_, BCAT);

    // Xin = bf16(concat(h, embed, edge))  [MROW x 640]
    {
        long long total = (long long)MROW * D1IN;
        pack_x_kernel<<<(unsigned)((total + 255) / 256), 256, 0, stream>>>(h_in, embed, edge, XIN, total);
    }

    // Z = relu(Xin @ el1^T + b1)            [MROW x 512]
    gemm_wmma_bf16<true, true><<<dim3(512 / 128, MROW / 128), 256, 0, stream>>>(XIN, EL1T, el1_b, Z, MROW, 512, D1IN);
    // H2 = Z @ el2^T + b2                   [MROW x 512]
    gemm_wmma_bf16<false, true><<<dim3(512 / 128, MROW / 128), 256, 0, stream>>>(Z, EL2T, el2_b, H2, MROW, 512, 512);
    // attention + child reductions
    attn_kernel<<<NNODE, 256, 0, stream>>>(H2, c_in, mask, att_w, att_b, HBAR, CSUM);
    // S1 = hbar @ nl1^T + b_nl1             [NNODE x 512]
    gemm_wmma_bf16<false, true><<<dim3(512 / 128, NNODE / 128), 256, 0, stream>>>(HBAR, NL1T, nl1_b, S1, NNODE, 512, 512);
    // S2 = S1 @ nl2^T + b_nl2               [NNODE x 512]
    gemm_wmma_bf16<false, true><<<dim3(512 / 128, NNODE / 128), 256, 0, stream>>>(S1, NL2T, nl2_b, S2, NNODE, 512, 512);
    // G = S2 @ [Wf|Wi|Wu|Wo]^T + bias_cat   [NNODE x 1024] f32
    gemm_wmma_bf16<false, false><<<dim3(1024 / 128, NNODE / 128), 256, 0, stream>>>(S2, WCAT, BCAT, G, NNODE, 1024, 512);
    // gates -> h_new, c_new
    final_kernel<<<(NNODE * 256 + 255) / 256, 256, 0, stream>>>(G, CSUM, (float*)d_out, NNODE * 256);

    (void)in_sizes; (void)n_in; (void)out_size; (void)ws_size;
}
